// CausalSelfAttention_13477607375822
// MI455X (gfx1250) — matrix operand order repaired and checked by reading, not compiled, not run on hardware
//
#include <hip/hip_runtime.h>
#include <hip/hip_bf16.h>
#include <stdint.h>

#define N_EMBD   1024
#define N_HEAD   16
#define HEAD_DIM 64
#define BATCH    2
#define SEQ      2048
#define C3       (3 * N_EMBD)

typedef __attribute__((ext_vector_type(16))) __bf16 v16bf;
typedef __attribute__((ext_vector_type(8)))  float  v8f;
typedef __attribute__((ext_vector_type(4)))  unsigned int u32x4;
typedef __attribute__((ext_vector_type(8)))  int   i32x8;
typedef __attribute__((ext_vector_type(4)))  int   i32x4;

#if defined(__has_builtin)
#if __has_builtin(__builtin_amdgcn_tensor_load_to_lds) && __has_builtin(__builtin_amdgcn_s_wait_tensorcnt)
#define USE_TDM 1
#endif
#endif

union Frag { v16bf v; uint4 q[2]; };

static __device__ __forceinline__ unsigned short f32_to_bf16_rne(float f) {
  union { float f; unsigned u; } x; x.f = f;
  unsigned u = x.u;
  unsigned r = (u + 0x7FFFu + ((u >> 16) & 1u)) >> 16;
  return (unsigned short)r;
}

static __device__ __forceinline__ v8f v8f_zero() {
  v8f v = {0.f, 0.f, 0.f, 0.f, 0.f, 0.f, 0.f, 0.f};
  return v;
}

// reductions across the 16-lane half-wave that owns a C-matrix row
static __device__ __forceinline__ float hmax16(float v) {
  v = fmaxf(v, __shfl_xor(v, 1, 32));
  v = fmaxf(v, __shfl_xor(v, 2, 32));
  v = fmaxf(v, __shfl_xor(v, 4, 32));
  v = fmaxf(v, __shfl_xor(v, 8, 32));
  return v;
}
static __device__ __forceinline__ float hsum16(float v) {
  v += __shfl_xor(v, 1, 32);
  v += __shfl_xor(v, 2, 32);
  v += __shfl_xor(v, 4, 32);
  v += __shfl_xor(v, 8, 32);
  return v;
}

#ifdef USE_TDM
// ---------------------------------------------------------------------------
// TDM: issue a 2D bf16 tile load (tile_k x tile_rows) from a row-major
// [tensor_rows x row_stride] tensor into LDS, padding 4 DWORDs every
// 16 DWORDs so the LDS row stride is 80 bytes (40 halves), matching the
// WMMA fragment loaders.  D# packing per CDNA5 ISA 8.3/8.4.
// ---------------------------------------------------------------------------
static __device__ __forceinline__ void tdm_load_2d(
    unsigned lds_off, const unsigned short* gptr,
    unsigned tile_k, unsigned tile_rows,
    unsigned tensor_k, unsigned tensor_rows, unsigned row_stride) {
  unsigned long long ga = (unsigned long long)(uintptr_t)gptr;

  u32x4 g0;
  g0[0] = 1u;                                          // count=1, user D#
  g0[1] = lds_off;                                     // LDS byte address
  g0[2] = (unsigned)(ga & 0xFFFFFFFFu);                // global_addr[31:0]
  g0[3] = (unsigned)((ga >> 32) & 0x01FFFFFFu) | (2u << 30);  // [56:32] | type=2

  i32x8 g1;
  // data_size=1(2B) | pad_enable | pad_interval=3(16 DW) | pad_amount=3(4 DW)
  g1[0] = (int)((1u << 16) | (1u << 20) | (3u << 22) | (3u << 25));
  g1[1] = (int)((tensor_k & 0xFFFFu) << 16);                              // dim0[15:0]
  g1[2] = (int)(((tensor_k >> 16) & 0xFFFFu) | ((tensor_rows & 0xFFFFu) << 16));
  g1[3] = (int)(((tensor_rows >> 16) & 0xFFFFu) | ((tile_k & 0xFFFFu) << 16));
  g1[4] = (int)(tile_rows & 0xFFFFu);                                     // tile_dim1
  g1[5] = (int)row_stride;                                                // dim0_stride
  g1[6] = 0;
  g1[7] = 0;

  i32x4 z = {0, 0, 0, 0};
#if __clang_major__ >= 23
  i32x8 z8 = {0, 0, 0, 0, 0, 0, 0, 0};
  __builtin_amdgcn_tensor_load_to_lds(g0, g1, z, z, z8, 0);
#else
  __builtin_amdgcn_tensor_load_to_lds(g0, g1, z, z, 0);
#endif
}
#endif

// ---------------------------------------------------------------------------
// fp32 -> bf16 conversion (grid-stride)
// ---------------------------------------------------------------------------
__global__ __launch_bounds__(256) void cvt_f32_bf16(
    const float* __restrict__ src, unsigned short* __restrict__ dst, int n) {
  int stride = gridDim.x * blockDim.x;
  for (int i = blockIdx.x * blockDim.x + threadIdx.x; i < n; i += stride)
    dst[i] = f32_to_bf16_rne(src[i]);
}

// ---------------------------------------------------------------------------
// V transpose: vt[b][h][d][t] = qkv[b*T+t][2C + h*64 + d]   (bf16)
// ---------------------------------------------------------------------------
__global__ __launch_bounds__(256) void transpose_v(
    const unsigned short* __restrict__ qkv, unsigned short* __restrict__ vt) {
  int idx = blockIdx.x * blockDim.x + threadIdx.x;   // B*H*D*T = 2^22 threads
  int t = idx & (SEQ - 1);
  int d = (idx >> 11) & (HEAD_DIM - 1);
  int h = (idx >> 17) & (N_HEAD - 1);
  int b = idx >> 21;
  vt[idx] = qkv[(size_t)(b * SEQ + t) * C3 + 2 * N_EMBD + h * HEAD_DIM + d];
}

// ---------------------------------------------------------------------------
// GEMM: out[M,N] = A[M,K] * Bw[N,K]^T   (A,Bw bf16 row-major, f32 accumulate)
// block tile 128x64, 8 waves -> each wave 32x32 (2x2 WMMA tiles).
// A/B k-slices staged to LDS by the Tensor Data Mover, double-buffered.
// ---------------------------------------------------------------------------
template <bool OUT_BF16>
__global__ __launch_bounds__(256) void gemm_nt_wmma(
    const unsigned short* __restrict__ A,
    const unsigned short* __restrict__ Bw,
    void* __restrict__ out, int M, int N, int K) {
  __shared__ unsigned short sA[2][128 * 40];  // stride 40 halves (80B, 16B aligned)
  __shared__ unsigned short sB[2][64 * 40];

  const int tid     = threadIdx.x;
  const int lane    = tid & 31;
  const int wave    = tid >> 5;
  const int wr      = wave >> 1;   // 0..3 : wave row (32 rows each)
  const int wc      = wave & 1;    // 0..1 : wave col (32 cols each)
  const int laneRow = lane & 15;
  const int hi      = lane >> 4;   // which half-wave

  const int blockN = blockIdx.x * 64;
  const int blockM = blockIdx.y * 128;

  v8f acc[2][2];
  acc[0][0] = v8f_zero(); acc[0][1] = v8f_zero();
  acc[1][0] = v8f_zero(); acc[1][1] = v8f_zero();

  const int nIter = K >> 5;

#ifdef USE_TDM
  const unsigned ldsA[2] = {(unsigned)(uintptr_t)&sA[0][0],
                            (unsigned)(uintptr_t)&sA[1][0]};
  const unsigned ldsB[2] = {(unsigned)(uintptr_t)&sB[0][0],
                            (unsigned)(uintptr_t)&sB[1][0]};
  if (wave == 0) {
    tdm_load_2d(ldsA[0], A + (size_t)blockM * K, 32, 128, (unsigned)K, (unsigned)M, (unsigned)K);
    tdm_load_2d(ldsB[0], Bw + (size_t)blockN * K, 32, 64, (unsigned)K, (unsigned)N, (unsigned)K);
  }
#endif

  for (int it = 0; it < nIter; ++it) {
    const int buf = it & 1;
#ifdef USE_TDM
    if (wave == 0) __builtin_amdgcn_s_wait_tensorcnt(0);
    __syncthreads();   // tile `buf` visible to all waves; previous compute done
    if (wave == 0 && it + 1 < nIter) {
      const int k0n = (it + 1) * 32;
      tdm_load_2d(ldsA[buf ^ 1], A + (size_t)blockM * K + k0n, 32, 128,
                  (unsigned)K, (unsigned)M, (unsigned)K);
      tdm_load_2d(ldsB[buf ^ 1], Bw + (size_t)blockN * K + k0n, 32, 64,
                  (unsigned)K, (unsigned)N, (unsigned)K);
    }
#else
    {  // cooperative fallback staging
      const int k0 = it * 32;
      for (int c = tid; c < 512; c += 256) {
        int row = c >> 2, kc = c & 3;
        const uint4* src = reinterpret_cast<const uint4*>(
            A + (size_t)(blockM + row) * K + k0 + kc * 8);
        *reinterpret_cast<uint4*>(&sA[buf][row * 40 + kc * 8]) = *src;
      }
      int row = tid >> 2, kc = tid & 3;
      const uint4* src = reinterpret_cast<const uint4*>(
          Bw + (size_t)(blockN + row) * K + k0 + kc * 8);
      *reinterpret_cast<uint4*>(&sB[buf][row * 40 + kc * 8]) = *src;
    }
    __syncthreads();
#endif

    const unsigned short* tA = &sA[buf][0];
    const unsigned short* tB = &sB[buf][0];

    // A-frag: lane<16 -> row=lane, K {0..7,16..23}; lane>=16 -> K {8..15,24..31}
    Frag a[2], b[2];
    #pragma unroll
    for (int mt = 0; mt < 2; ++mt) {
      int r  = wr * 32 + mt * 16 + laneRow;
      int ka = hi ? 8 : 0;
      a[mt].q[0] = *reinterpret_cast<const uint4*>(&tA[r * 40 + ka]);
      a[mt].q[1] = *reinterpret_cast<const uint4*>(&tA[r * 40 + ka + 16]);
    }
    // B-frag: lane<16 -> col=lane, K {0..7,16..23}; lane>=16 -> K {8..15,24..31}
    #pragma unroll
    for (int nt = 0; nt < 2; ++nt) {
      int r  = wc * 32 + nt * 16 + laneRow;
      int kb = hi ? 8 : 0;
      b[nt].q[0] = *reinterpret_cast<const uint4*>(&tB[r * 40 + kb]);
      b[nt].q[1] = *reinterpret_cast<const uint4*>(&tB[r * 40 + kb + 16]);
    }
    #pragma unroll
    for (int mt = 0; mt < 2; ++mt)
      #pragma unroll
      for (int nt = 0; nt < 2; ++nt)
        acc[mt][nt] = __builtin_amdgcn_wmma_f32_16x16x32_bf16(
            false, a[mt].v, false, b[nt].v, (short)0, acc[mt][nt], false, false);
    __syncthreads();
  }

  // epilogue: C layout -> VGPR i, lane: row = i + hi*8, col = laneRow
  #pragma unroll
  for (int mt = 0; mt < 2; ++mt)
    #pragma unroll
    for (int nt = 0; nt < 2; ++nt)
      #pragma unroll
      for (int i = 0; i < 8; ++i) {
        int row = blockM + wr * 32 + mt * 16 + i + hi * 8;
        int col = blockN + wc * 32 + nt * 16 + laneRow;
        if (OUT_BF16)
          ((unsigned short*)out)[(size_t)row * N + col] = f32_to_bf16_rne(acc[mt][nt][i]);
        else
          ((float*)out)[(size_t)row * N + col] = acc[mt][nt][i];
      }
}

// ---------------------------------------------------------------------------
// Causal flash attention: one block per (b*H + h, 64 query rows); 4 waves,
// each wave owns 16 query rows. Streams 32-key blocks with online softmax.
// ---------------------------------------------------------------------------
__global__ __launch_bounds__(128) void flash_attn_wmma(
    const unsigned short* __restrict__ qkv,  // [B*T, 3C] bf16
    const unsigned short* __restrict__ vt,   // [B,H,D,T] bf16
    unsigned short* __restrict__ y)          // [B*T, C]  bf16
{
  __shared__ unsigned short sP[4][16 * 40];  // per-wave P repack scratch

  const int lane    = threadIdx.x & 31;
  const int wave    = threadIdx.x >> 5;
  const int laneRow = lane & 15;
  const int hi      = lane >> 4;
  const int bh      = blockIdx.y;            // b*H + h
  const int b       = bh >> 4;
  const int h       = bh & 15;
  const int q0      = blockIdx.x * 64 + wave * 16;

  // Q A-fragments for the two 32-wide d chunks (D=64)
  Frag aq[2];
  {
    const int t = q0 + laneRow;
    const size_t base = (size_t)(b * SEQ + t) * C3 + h * HEAD_DIM;
    #pragma unroll
    for (int dt = 0; dt < 2; ++dt) {
      int k0 = dt * 32 + (hi ? 8 : 0);
      aq[dt].q[0] = *reinterpret_cast<const uint4*>(qkv + base + k0);
      aq[dt].q[1] = *reinterpret_cast<const uint4*>(qkv + base + k0 + 16);
    }
  }

  v8f o[4];
  o[0] = v8f_zero(); o[1] = v8f_zero(); o[2] = v8f_zero(); o[3] = v8f_zero();
  float mrow[8], lrow[8];
  #pragma unroll
  for (int i = 0; i < 8; ++i) { mrow[i] = -1e30f; lrow[i] = 0.f; }

  const float scale = 0.125f;  // 1/sqrt(64)
  const int nkb = (q0 + 16 + 31) / 32;

  for (int kb = 0; kb < nkb; ++kb) {
    const int t0 = kb * 32;

    // S(16x32) = Q(16x64) * K^T : two 16x16 C tiles, 2 WMMA each
    v8f s[2];
    s[0] = v8f_zero(); s[1] = v8f_zero();
    #pragma unroll
    for (int ct = 0; ct < 2; ++ct)
      #pragma unroll
      for (int dt = 0; dt < 2; ++dt) {
        Frag bk;
        int key = t0 + ct * 16 + laneRow;
        int d0  = dt * 32 + (hi ? 8 : 0);
        const size_t ka = (size_t)(b * SEQ + key) * C3 + N_EMBD + h * HEAD_DIM + d0;
        bk.q[0] = *reinterpret_cast<const uint4*>(qkv + ka);
        bk.q[1] = *reinterpret_cast<const uint4*>(qkv + ka + 16);
        s[ct] = __builtin_amdgcn_wmma_f32_16x16x32_bf16(
            false, aq[dt].v, false, bk.v, (short)0, s[ct], false, false);
      }

    // scale + causal mask + online softmax (row = i + hi*8 within wave tile)
    #pragma unroll
    for (int i = 0; i < 8; ++i) {
      const int qt   = q0 + i + hi * 8;
      const int key0 = t0 + laneRow;
      float a0 = s[0][i] * scale;
      float a1 = s[1][i] * scale;
      if (key0 > qt)      a0 = -1e30f;
      if (key0 + 16 > qt) a1 = -1e30f;
      float mx = hmax16(fmaxf(a0, a1));
      mx = fmaxf(mx, mrow[i]);
      float f  = __expf(mrow[i] - mx);
      float p0 = __expf(a0 - mx);
      float p1 = __expf(a1 - mx);
      float rs = hsum16(p0 + p1);
      lrow[i] = lrow[i] * f + rs;
      mrow[i] = mx;
      o[0][i] *= f; o[1][i] *= f; o[2][i] *= f; o[3][i] *= f;
      const int prow = i + hi * 8;
      sP[wave][prow * 40 + laneRow]      = f32_to_bf16_rne(p0);
      sP[wave][prow * 40 + 16 + laneRow] = f32_to_bf16_rne(p1);
    }
    __builtin_amdgcn_fence(__ATOMIC_ACQ_REL, "wavefront");

    // repack P (C layout in LDS) into A-fragment layout
    Frag ap;
    {
      int ka = hi ? 8 : 0;
      ap.q[0] = *reinterpret_cast<const uint4*>(&sP[wave][laneRow * 40 + ka]);
      ap.q[1] = *reinterpret_cast<const uint4*>(&sP[wave][laneRow * 40 + ka + 16]);
    }

    // O(16x64) += P(16x32) * V(32x64): 4 N tiles; Vt gives contiguous K loads
    #pragma unroll
    for (int nt = 0; nt < 4; ++nt) {
      Frag bv;
      int d = nt * 16 + laneRow;
      const size_t va = ((size_t)(bh * HEAD_DIM + d)) * SEQ + t0 + (hi ? 8 : 0);
      bv.q[0] = *reinterpret_cast<const uint4*>(vt + va);
      bv.q[1] = *reinterpret_cast<const uint4*>(vt + va + 16);
      o[nt] = __builtin_amdgcn_wmma_f32_16x16x32_bf16(
          false, ap.v, false, bv.v, (short)0, o[nt], false, false);
    }
  }

  // normalize and write y (bf16)
  #pragma unroll
  for (int i = 0; i < 8; ++i) {
    const float inv = 1.0f / lrow[i];
    const int t = q0 + i + hi * 8;
    const size_t rb = (size_t)(b * SEQ + t) * N_EMBD + h * HEAD_DIM;
    #pragma unroll
    for (int nt = 0; nt < 4; ++nt)
      y[rb + nt * 16 + laneRow] = f32_to_bf16_rne(o[nt][i] * inv);
  }
}

// ---------------------------------------------------------------------------
// Launcher
// ---------------------------------------------------------------------------
extern "C" void kernel_launch(void* const* d_in, const int* in_sizes, int n_in,
                              void* d_out, int out_size, void* d_ws, size_t ws_size,
                              hipStream_t stream) {
  const float* x  = (const float*)d_in[0];   // [B,T,C]
  const float* Wa = (const float*)d_in[1];   // [3C,C]
  const float* Wp = (const float*)d_in[2];   // [C,C]
  float* out = (float*)d_out;                // [B,T,C]

  // workspace layout (bf16 halves)
  unsigned short* w = (unsigned short*)d_ws;
  unsigned short* xb  = w;                   //  4,194,304
  unsigned short* wab = w + 4194304;         //  3,145,728
  unsigned short* wpb = w + 7340032;         //  1,048,576
  unsigned short* qkv = w + 8388608;         // 12,582,912
  unsigned short* vt  = w + 20971520;        //  4,194,304
  unsigned short* yb  = w + 25165824;        //  4,194,304
  if (ws_size < (size_t)29360128 * 2) return;

  cvt_f32_bf16<<<4096, 256, 0, stream>>>(x,  xb,  BATCH * SEQ * N_EMBD);
  cvt_f32_bf16<<<4096, 256, 0, stream>>>(Wa, wab, C3 * N_EMBD);
  cvt_f32_bf16<<<2048, 256, 0, stream>>>(Wp, wpb, N_EMBD * N_EMBD);

  // qkv = x @ W_attn^T
  gemm_nt_wmma<true><<<dim3(C3 / 64, (BATCH * SEQ) / 128), 256, 0, stream>>>(
      xb, wab, qkv, BATCH * SEQ, C3, N_EMBD);

  transpose_v<<<(BATCH * N_HEAD * HEAD_DIM * SEQ) / 256, 256, 0, stream>>>(qkv, vt);

  flash_attn_wmma<<<dim3(SEQ / 64, BATCH * N_HEAD), 128, 0, stream>>>(qkv, vt, yb);

  // out = y @ W_proj^T  (f32 output)
  gemm_nt_wmma<false><<<dim3(N_EMBD / 64, (BATCH * SEQ) / 128), 256, 0, stream>>>(
      yb, wpb, out, BATCH * SEQ, N_EMBD, N_EMBD);
}
